// MyLSTM_39900246180375
// MI455X (gfx1250) — matrix operand order repaired and checked by reading, not compiled, not run on hardware
//
#include <hip/hip_runtime.h>
#include <hip/hip_bf16.h>

// ---------------------------------------------------------------------------
// Persistent fused LSTM for MI455X (gfx1250, wave32, WMMA).
//   z_t = [x_t ; h_{t-1}] @ [W ; R] + bias      (K = 1024, bf16 WMMA, f32 acc)
//   c_t = tanh(f*c + i*g);  h_t = o*c_t         (native v_tanh/v_exp, branch-free)
// 32 persistent blocks, each owns 16 hidden columns (x4 gates = 64 N cols),
// M = 128 batch rows, device-wide sense-reversing barrier per timestep.
// Fragment loads are explicitly double-buffered so LDS latency overlaps the
// matrix pipe instead of forcing s_wait_dscnt 0 before every WMMA.
// ---------------------------------------------------------------------------

typedef __attribute__((ext_vector_type(16))) __bf16 v16bf;
typedef __attribute__((ext_vector_type(8)))  __bf16 v8bf;
typedef __attribute__((ext_vector_type(8)))  float  v8f;

constexpr int Bdim = 128;
constexpr int Tdim = 1024;
constexpr int Idim = 512;
constexpr int Hdim = 512;
constexpr int G4   = 4 * Hdim;        // 2048 gate columns
constexpr int NBLK = 32;              // persistent blocks (all resident)
constexpr int HS   = Hdim / NBLK;     // 16 hidden cols per block
constexpr int KTOT = Idim + Hdim;     // 1024 fused K ([x ; h])
constexpr int KC   = 32;              // K per v_wmma_f32_16x16x32_bf16
constexpr int NKC  = KTOT / KC;       // 32 wmma K-chunks
constexpr int KSTG = 128;             // K per staging super-chunk (4 wmma chunks)
constexpr int NSTG = KTOT / KSTG;     // 8 staging stages (0-3: x, 4-7: h)
constexpr int AST  = 136;             // sA row stride in bf16 (272 B: 16B-aligned, bank-spread)

__device__ __forceinline__ float fast_exp2(float x) { return __builtin_amdgcn_exp2f(x); }
__device__ __forceinline__ float fast_rcp (float x) { return __builtin_amdgcn_rcpf(x); }

__device__ __forceinline__ float fast_sigmoid(float x) {
    // 1/(1+exp(-x)) with raw v_exp_f32 + v_rcp_f32 (no libm branches)
    return fast_rcp(1.0f + fast_exp2(-1.4426950408889634f * x));
}

__device__ __forceinline__ float fast_tanh(float x) {
#if __has_builtin(__builtin_amdgcn_tanhf)
    return __builtin_amdgcn_tanhf(x);            // native v_tanh_f32 on gfx1250
#else
    // tanh(x) = 1 - 2/(exp2(2*log2e*x)+1); saturates correctly via inf/0
    const float e = fast_exp2(2.885390081777927f * x);
    return 1.0f - 2.0f * fast_rcp(e + 1.0f);
#endif
}

// Re-initialize barrier + h0 every launch (d_ws is poisoned by the harness;
// launches must be stateless/deterministic for graph replay).
__global__ void lstm_init_kernel(unsigned* __restrict__ bar,
                                 float* __restrict__ h0, int n) {
    int i = blockIdx.x * blockDim.x + threadIdx.x;
    if (i < 2) bar[i] = 0u;
    for (int j = i; j < n; j += gridDim.x * blockDim.x) h0[j] = 0.0f;
}

__global__ __launch_bounds__(256)
void lstm_persistent_kernel(const float* __restrict__ x,     // [B,T,I]
                            const float* __restrict__ Wk,    // [I,4H]
                            const float* __restrict__ Rk,    // [H,4H]
                            const float* __restrict__ bias,  // [4H]
                            float* __restrict__ out,         // [B,T,H]
                            unsigned* __restrict__ barCnt,
                            unsigned* __restrict__ barGen,
                            float* __restrict__ hbuf0,       // [B,H] ping
                            float* __restrict__ hbuf1)       // [B,H] pong
{
    // sB: fused [W;R] slice, pre-arranged in WMMA B-fragment order (bf16):
    //   index ((kc*4 + nt)*32 + lane)*16 + e
    //   lane<16: N=lane,    K = kc*32 + (e<8 ? e : 8 + e)      (e = 0..15)
    //   lane>=16:N=lane-16, K = kc*32 + 8 + (e<8 ? e : 8 + e)
    __shared__ __bf16 sB[NKC * 4 * 32 * 16];            // 128 KB
    // sA: double-buffered staging of one 128-K super-chunk of [x_t ; h]
    __shared__ __bf16 sA[2][Bdim * AST];                // 2 x 34 KB

    const int tid  = threadIdx.x;
    const int lane = tid & 31;
    const int wv   = tid >> 5;            // wave id = M-tile (16 rows each)
    const int hoff = blockIdx.x * HS;     // hidden-column slice owned by block
    const int colL = lane & 15;           // local N within 16-wide tile
    const int kb   = (lane < 16) ? 0 : 8; // A-fragment K-half select

    // ---- one-time: build bf16 B-fragments of [W;R] slice in LDS ----------
    for (int idx = tid; idx < NKC * 4 * 32; idx += blockDim.x) {
        const int kc = idx >> 7;           // / (4*32)
        const int nt = (idx >> 5) & 3;     // gate
        const int ln = idx & 31;
        const int col   = nt * Hdim + hoff + (ln & 15);
        const int kbase = kc * KC + ((ln < 16) ? 0 : 8);
        __bf16* dst = &sB[idx * 16];
#pragma unroll
        for (int e = 0; e < 16; ++e) {
            const int kg = kbase + ((e < 8) ? e : e + 8);
            const float v = (kg < Idim) ? Wk[(size_t)kg * G4 + col]
                                        : Rk[(size_t)(kg - Idim) * G4 + col];
            dst[e] = (__bf16)v;
        }
    }

    // per-lane bias for each gate (depends only on N column)
    float bgate[4];
#pragma unroll
    for (int nt = 0; nt < 4; ++nt) bgate[nt] = bias[nt * Hdim + hoff + colL];

    __syncthreads();

    // cell state: 8 rows per lane (C-matrix layout), persistent in registers
    float creg[8];
#pragma unroll
    for (int r = 0; r < 8; ++r) creg[r] = 0.0f;

    const float* hread  = hbuf0;   // zeroed h0
    float*       hwrite = hbuf1;

    const int srow = tid >> 1;     // staging: batch row handled by this thread
    const int sseg = tid & 1;      // staging: 64-wide K segment within stage

    for (int t = 0; t < Tdim; ++t) {
        // stage 's' (128 K of [x_t ; h_{t-1}], bf16) into buffer 'buf'
        auto stage = [&](int s, __bf16* buf) {
            const int k0 = s * KSTG + sseg * 64;
            const float* src = (s < NSTG / 2)
                ? &x[(size_t)srow * (Tdim * Idim) + (size_t)t * Idim + k0]
                : &hread[(size_t)srow * Hdim + (k0 - Idim)];
            __bf16* dst = &buf[srow * AST + sseg * 64];
#pragma unroll
            for (int j = 0; j < 16; ++j) {
                const float4 v = ((const float4*)src)[j];
                dst[4 * j + 0] = (__bf16)v.x;
                dst[4 * j + 1] = (__bf16)v.y;
                dst[4 * j + 2] = (__bf16)v.z;
                dst[4 * j + 3] = (__bf16)v.w;
            }
        };

        // accumulators start at bias (z = bias + [x;h]@[W;R])
        v8f acc[4];
#pragma unroll
        for (int nt = 0; nt < 4; ++nt)
#pragma unroll
            for (int r = 0; r < 8; ++r) acc[nt][r] = bgate[nt];

        stage(0, sA[0]);
        for (int s = 0; s < NSTG; ++s) {
            __syncthreads();                       // stage(s) visible; buf reuse safe
            if (s + 1 < NSTG) stage(s + 1, sA[(s + 1) & 1]);

            const __bf16* abuf = sA[s & 1];

            // explicit double-buffered fragment pipeline: load frags(kcc+1)
            // while the matrix pipe chews on frags(kcc).
            v16bf afr[2];
            v16bf bfr[2][4];
            auto loadFrags = [&](int kcc, int pb) {
                const __bf16* ap =
                    &abuf[(wv * 16 + colL) * AST + kcc * KC + kb];
                const v8bf alo = *(const v8bf*)(ap);       // K = kb..kb+7
                const v8bf ahi = *(const v8bf*)(ap + 16);  // K = kb+16..kb+23
#pragma unroll
                for (int j = 0; j < 8; ++j) {
                    afr[pb][j] = alo[j];
                    afr[pb][8 + j] = ahi[j];
                }
                const int kc = (s << 2) + kcc;
#pragma unroll
                for (int nt = 0; nt < 4; ++nt)
                    bfr[pb][nt] =
                        *(const v16bf*)&sB[(((kc << 2) + nt) * 32 + lane) * 16];
            };

            loadFrags(0, 0);
#pragma unroll
            for (int kcc = 0; kcc < 4; ++kcc) {    // 4 wmma K-chunks per stage
                if (kcc + 1 < 4) loadFrags(kcc + 1, (kcc + 1) & 1);
                const int pb = kcc & 1;
#pragma unroll
                for (int nt = 0; nt < 4; ++nt) {
                    acc[nt] = __builtin_amdgcn_wmma_f32_16x16x32_bf16(
                        false, afr[pb], false, bfr[pb][nt], (short)0, acc[nt],
                        false, false);
                }
            }
        }

        // ---- elementwise LSTM update + stores (C layout: VGPR r, lane) ----
#pragma unroll
        for (int r = 0; r < 8; ++r) {
            const float it = fast_sigmoid(acc[0][r]);
            const float ft = fast_sigmoid(acc[1][r]);
            const float gt = fast_tanh(acc[2][r]);
            const float ot = fast_sigmoid(acc[3][r]);
            const float cn = fast_tanh(__builtin_fmaf(ft, creg[r], it * gt));
            creg[r] = cn;                           // reference carries tanh'd cell
            const float hn = ot * cn;
            const int brow = wv * 16 + ((lane < 16) ? r : r + 8);
            out[(size_t)brow * (Tdim * Hdim) + (size_t)t * Hdim + hoff + colL] = hn;
            hwrite[(size_t)brow * Hdim + hoff + colL] = hn;
        }

        // prefetch next timestep's x rows into caches (global_prefetch_b8)
        if (t + 1 < Tdim) {
            __builtin_prefetch(
                &x[(size_t)srow * (Tdim * Idim) + (size_t)(t + 1) * Idim + sseg * 64],
                0, 1);
        }

        // swap h ping-pong
        { const float* tr = hread; hread = hwrite; hwrite = (float*)tr; }

        // ---- device-wide sense-reversing barrier ----
        __threadfence();
        if (tid == 0) {
            const unsigned g =
                __hip_atomic_load(barGen, __ATOMIC_RELAXED, __HIP_MEMORY_SCOPE_AGENT);
            const unsigned a =
                __hip_atomic_fetch_add(barCnt, 1u, __ATOMIC_ACQ_REL, __HIP_MEMORY_SCOPE_AGENT);
            if (a == (unsigned)(NBLK - 1)) {
                __hip_atomic_store(barCnt, 0u, __ATOMIC_RELAXED, __HIP_MEMORY_SCOPE_AGENT);
                __hip_atomic_fetch_add(barGen, 1u, __ATOMIC_ACQ_REL, __HIP_MEMORY_SCOPE_AGENT);
            } else {
                while (__hip_atomic_load(barGen, __ATOMIC_ACQUIRE,
                                         __HIP_MEMORY_SCOPE_AGENT) == g)
                    __builtin_amdgcn_s_sleep(2);
            }
        }
        __syncthreads();
        __threadfence();
    }
}

extern "C" void kernel_launch(void* const* d_in, const int* in_sizes, int n_in,
                              void* d_out, int out_size, void* d_ws, size_t ws_size,
                              hipStream_t stream) {
    const float* x    = (const float*)d_in[0];   // [B,T,I]
    const float* Wk   = (const float*)d_in[1];   // [I,4H]
    const float* Rk   = (const float*)d_in[2];   // [H,4H]
    const float* bias = (const float*)d_in[3];   // [4H]
    float* out = (float*)d_out;                  // [B,T,H]

    unsigned* bar = (unsigned*)d_ws;             // [cnt, gen]
    float* hb0 = (float*)d_ws + 64;              // [B,H] ping (256 KB)
    float* hb1 = hb0 + Bdim * Hdim;              // [B,H] pong

    lstm_init_kernel<<<64, 256, 0, stream>>>(bar, hb0, Bdim * Hdim);
    lstm_persistent_kernel<<<NBLK, 256, 0, stream>>>(
        x, Wk, Rk, bias, out, bar, bar + 1, hb0, hb1);
}
